// Posit4LinearLayer_66176856097493
// MI455X (gfx1250) — matrix operand order repaired and checked by reading, not compiled, not run on hardware
//
#include <hip/hip_runtime.h>

typedef __attribute__((ext_vector_type(16))) _Float16 v16h;
typedef __attribute__((ext_vector_type(8)))  _Float16 v8h;
typedef __attribute__((ext_vector_type(4)))  _Float16 v4h;
typedef __attribute__((ext_vector_type(8)))  float    v8f;

#define KDIM 4096
#define NDIM 11008
#define MDIM 32
#define KC   512            // K-chunk per block (8 chunks total)
#define LS   (KC + 8)       // LDS row stride in halves: 520 -> 260 dwords == 4 mod 64 (bank friendly)
#define QSCALE (32.0f / 15.0f)

// ---------------- kernel 1: t[b] = sum_k x[b,k] ----------------
__global__ __launch_bounds__(1024)
void rowsum_kernel(const float* __restrict__ x, float* __restrict__ t) {
    const int row  = threadIdx.x >> 5;   // 32 waves, one row each
    const int lane = threadIdx.x & 31;
    const float* rp = x + (size_t)row * KDIM;
    float s = 0.0f;
    for (int k = lane; k < KDIM; k += 32) s += rp[k];
    #pragma unroll
    for (int off = 16; off > 0; off >>= 1) s += __shfl_xor(s, off, 32);
    if (lane == 0) t[row] = s;
}

// -------- kernel 2: out[b,o] = dequant(bias[o]) - 16 * t[b] --------
__global__ __launch_bounds__(256)
void init_out_kernel(const int* __restrict__ qbias, const float* __restrict__ t,
                     float* __restrict__ out) {
    const int idx = blockIdx.x * 256 + threadIdx.x;
    if (idx >= MDIM * NDIM) return;
    const int b = idx / NDIM;
    const int o = idx - b * NDIM;
    out[idx] = ((float)qbias[o] * QSCALE - 16.0f) - 16.0f * t[b];
}

// -------- kernel 3: out[b,o] += (32/15) * sum_k x_f16[b,k]*code[o,k] --------
__global__ __launch_bounds__(256)
void posit4_gemm_kernel(const float* __restrict__ x, const int* __restrict__ w,
                        float* __restrict__ out) {
    __shared__ _Float16 sx[MDIM * LS];   // 16,640 halves = 33.3 KB

    const int tid = threadIdx.x;
    const int kc0 = blockIdx.y * KC;

    // Stage x[:, kc0:kc0+KC] into LDS as f16: float4 loads, 4xf16 (b64) stores.
    for (int i = tid; i < (MDIM * KC) / 4; i += 256) {
        const int r = i >> 7;              // KC/4 == 128 vectors per row
        const int c = (i & 127) << 2;
        const float4 f = *(const float4*)(x + (size_t)r * KDIM + kc0 + c);
        v4h h;
        h[0] = (_Float16)f.x; h[1] = (_Float16)f.y;
        h[2] = (_Float16)f.z; h[3] = (_Float16)f.w;
        *(v4h*)(sx + r * LS + c) = h;
    }
    __syncthreads();

    const int wave  = tid >> 5;
    const int lane  = tid & 31;
    const int n     = lane & 15;         // column within N-tile / A-matrix row
    const int hi    = lane >> 4;         // lane half selects K sub-range
    const int obase = (blockIdx.x * 8 + wave) * 16;

    // B fragment (32x16 f16): lane = output feature column; lanes 0-15 hold
    // K=kb+0..15, lanes 16-31 hold K=kb+16..31 -> 16 consecutive int32 codes.
    const int* wp = w + (size_t)(obase + n) * KDIM + kc0 + hi * 16;

    // A fragment (16x32 f16) from LDS: lanes 0-15: row n, K 0..15;
    // lanes 16-31: row n, K 16..31.
    const _Float16* a0p = sx + n * LS + hi * 16;         // M-tile 0 (rows 0..15)
    const _Float16* a1p = a0p + 16 * LS;                 // M-tile 1 (rows 16..31)

    v8f acc0 = {};
    v8f acc1 = {};

    #pragma unroll 4
    for (int kb = 0; kb < KC; kb += 32) {
        const int4* q = (const int4*)wp;
        const int4 q0 = q[0];
        const int4 q1 = q[1];
        const int4 q2 = q[2];
        const int4 q3 = q[3];
        __builtin_prefetch((const void*)(wp + 128), 0, 0);   // global_prefetch_b8: pull stream into GL2
        wp += 32;

        v16h bf;
        bf[0]  = (_Float16)q0.x; bf[1]  = (_Float16)q0.y;
        bf[2]  = (_Float16)q0.z; bf[3]  = (_Float16)q0.w;
        bf[4]  = (_Float16)q1.x; bf[5]  = (_Float16)q1.y;
        bf[6]  = (_Float16)q1.z; bf[7]  = (_Float16)q1.w;
        bf[8]  = (_Float16)q2.x; bf[9]  = (_Float16)q2.y;
        bf[10] = (_Float16)q2.z; bf[11] = (_Float16)q2.w;
        bf[12] = (_Float16)q3.x; bf[13] = (_Float16)q3.y;
        bf[14] = (_Float16)q3.z; bf[15] = (_Float16)q3.w;

        const v8h al0 = *(const v8h*)(a0p + kb);
        const v8h ah0 = *(const v8h*)(a0p + kb + 8);
        const v16h af0 = __builtin_shufflevector(al0, ah0,
            0, 1, 2, 3, 4, 5, 6, 7, 8, 9, 10, 11, 12, 13, 14, 15);
        const v8h al1 = *(const v8h*)(a1p + kb);
        const v8h ah1 = *(const v8h*)(a1p + kb + 8);
        const v16h af1 = __builtin_shufflevector(al1, ah1,
            0, 1, 2, 3, 4, 5, 6, 7, 8, 9, 10, 11, 12, 13, 14, 15);

        acc0 = __builtin_amdgcn_wmma_f32_16x16x32_f16(
            false, af0, false, bf, (short)0, acc0, false, false);
        acc1 = __builtin_amdgcn_wmma_f32_16x16x32_f16(
            false, af1, false, bf, (short)0, acc1, false, false);
    }

    // C/D layout: VGPR r, lanes 0-15 -> M=r, lanes 16-31 -> M=r+8; N = lane&15.
    float* op = out + obase + n;
    #pragma unroll
    for (int r = 0; r < 8; ++r) {
        const int m = r + hi * 8;
        __hip_atomic_fetch_add(op + (size_t)m * NDIM, QSCALE * acc0[r],
                               __ATOMIC_RELAXED, __HIP_MEMORY_SCOPE_AGENT);
        __hip_atomic_fetch_add(op + (size_t)(m + 16) * NDIM, QSCALE * acc1[r],
                               __ATOMIC_RELAXED, __HIP_MEMORY_SCOPE_AGENT);
    }
}

extern "C" void kernel_launch(void* const* d_in, const int* in_sizes, int n_in,
                              void* d_out, int out_size, void* d_ws, size_t ws_size,
                              hipStream_t stream) {
    const float* x  = (const float*)d_in[0];   // [32, 4096] f32
    const int*   qw = (const int*)d_in[1];     // [11008, 4096] int32 codes
    const int*   qb = (const int*)d_in[2];     // [11008] int32 codes
    float* out = (float*)d_out;                // [32, 11008] f32
    float* t   = (float*)d_ws;                 // 32 floats of scratch

    rowsum_kernel<<<1, 1024, 0, stream>>>(x, t);

    const int outElems = MDIM * NDIM;
    init_out_kernel<<<(outElems + 255) / 256, 256, 0, stream>>>(qb, t, out);

    dim3 grid(NDIM / (8 * 16), KDIM / KC);     // (86, 8)
    posit4_gemm_kernel<<<grid, 256, 0, stream>>>(x, qw, out);
}
